// PGCU_54485955117218
// MI455X (gfx1250) — matrix operand order repaired and checked by reading, not compiled, not run on hardware
//
#include <hip/hip_runtime.h>
#include <hip/hip_bf16.h>

// ---------------------------------------------------------------------------
// PGCU fused attention-upsampler for MI455X (gfx1250, wave32, WMMA).
//
// Pipeline:
//   k_fold : Wfold[128][96](f16) = fl_w @ f_w  (linear folded into conv), bias'
//   k_ds x10: DownSamplingBlocks for G/V key paths (tiny, /64 resolution)
//   k_gkv  : G conv + Gk = LN(linear)  (f16, WMMA-B layout) ; V conv
//   k_main : per 16x16 pixel tile:
//              c8 (fm(up_x)||fp(guide)) staged in LDS with halo,
//              folded conv via v_wmma_f32_16x16x32_f16 (24 WMMA / pixel-row),
//              LayerNorm, scores vs 64 keys via WMMA (16 WMMA / pixel-row),
//              softmax(64) + V-weighted sum -> ATT[4,512,512]
//   k_final: 3x3 conv fa -> d_out
// ---------------------------------------------------------------------------

typedef __attribute__((ext_vector_type(16))) _Float16 v16h;
typedef __attribute__((ext_vector_type(8)))  _Float16 v8h;
typedef __attribute__((ext_vector_type(8)))  float    v8f;

union V16 { v16h v; _Float16 h[16]; };

// ---------------------------------------------------------------------------
// Fold fl_w (4,32,128 -> 128x128) into f_w (128,8,3,3 -> 128x72).
// Wfold[o][k] = sum_v fl_w[o][v] * f_w[v][k], k padded 72->96 with zeros.
// bias'[o]    = sum_v fl_w[o][v] * f_b[v] + fl_b[o]
// ---------------------------------------------------------------------------
__global__ __launch_bounds__(256) void k_fold(
    const float* __restrict__ fl_w, const float* __restrict__ f_w,
    const float* __restrict__ f_b,  const float* __restrict__ fl_b,
    _Float16* __restrict__ WF, float* __restrict__ BIAS)
{
    int t = blockIdx.x * 256 + threadIdx.x;
    if (t >= 128 * 96) return;
    int o = t / 96, k = t % 96;
    float acc = 0.f;
    if (k < 72) {
        for (int v = 0; v < 128; ++v) acc += fl_w[o * 128 + v] * f_w[v * 72 + k];
    }
    WF[o * 96 + k] = (_Float16)acc;
    if (k == 0) {
        float b = fl_b[o];
        for (int v = 0; v < 128; ++v) b += fl_w[o * 128 + v] * f_b[v];
        BIAS[o] = b;
    }
}

// ---------------------------------------------------------------------------
// DownSamplingBlock: 3x3 conv stride 2 pad 1 (Cin -> 4) then 2x2 maxpool.
// out: [4][H/4][W/4]
// ---------------------------------------------------------------------------
__global__ __launch_bounds__(256) void k_ds(
    const float* __restrict__ in, int H, int W, int Cin,
    const float* __restrict__ w, const float* __restrict__ b,
    float* __restrict__ out)
{
    int OH = H >> 2, OW = W >> 2;
    int total = 4 * OH * OW;
    int t = blockIdx.x * 256 + threadIdx.x;
    if (t >= total) return;
    int co = t / (OH * OW);
    int r  = t % (OH * OW);
    int oy = r / OW, ox = r % OW;
    float mx = -3.0e38f;
    for (int py = 0; py < 2; ++py)
    for (int px = 0; px < 2; ++px) {
        int cy = 2 * oy + py, cx = 2 * ox + px;
        float acc = b[co];
        for (int ci = 0; ci < Cin; ++ci)
            for (int dy = 0; dy < 3; ++dy)
                for (int dx = 0; dx < 3; ++dx) {
                    int iy = 2 * cy + dy - 1, ix = 2 * cx + dx - 1;
                    if ((unsigned)iy < (unsigned)H && (unsigned)ix < (unsigned)W)
                        acc += w[(co * Cin + ci) * 9 + dy * 3 + dx] *
                               in[(ci * H + iy) * W + ix];
                }
        mx = fmaxf(mx, acc);
    }
    out[t] = mx;
}

// ---------------------------------------------------------------------------
// Key/value head: G = conv3x3(concat(Gm,Gp)) [128,8,8];
// Gk[c][m][d] = LN(gl_w @ G + gl_b) stored f16 as [4][64][32] (WMMA-B rows);
// V[c][m]    = conv3x3(concat(Vm,Vp)). Single block, 256 threads.
// ---------------------------------------------------------------------------
__global__ __launch_bounds__(256) void k_gkv(
    const float* __restrict__ gmm, const float* __restrict__ gpp,
    const float* __restrict__ vmm, const float* __restrict__ vpp,
    const float* __restrict__ g_w, const float* __restrict__ g_b,
    const float* __restrict__ gl_w, const float* __restrict__ gl_b,
    const float* __restrict__ gl_g, const float* __restrict__ gl_beta,
    const float* __restrict__ v_w, const float* __restrict__ v_b,
    _Float16* __restrict__ GK, float* __restrict__ VV)
{
    extern __shared__ char sm[];
    float* sG  = (float*)sm;             // [128][64]
    float* sK  = (float*)(sm + 32768);   // [64][128]
    float* sIn = (float*)(sm + 65536);   // [8][8][8]
    int tid = threadIdx.x;

    for (int i = tid; i < 512; i += 256) {
        int ch = i >> 6, pos = i & 63;
        sIn[i] = (ch < 4) ? gmm[ch * 64 + pos] : gpp[(ch - 4) * 64 + pos];
    }
    __syncthreads();
    for (int p = tid; p < 8192; p += 256) {             // G conv
        int v = p >> 6, m = p & 63, my = m >> 3, mx = m & 7;
        float acc = g_b[v];
        for (int ci = 0; ci < 8; ++ci)
            for (int dy = 0; dy < 3; ++dy)
                for (int dx = 0; dx < 3; ++dx) {
                    int iy = my + dy - 1, ix = mx + dx - 1;
                    if ((unsigned)iy < 8u && (unsigned)ix < 8u)
                        acc += g_w[(v * 8 + ci) * 9 + dy * 3 + dx] *
                               sIn[ci * 64 + iy * 8 + ix];
                }
        sG[p] = acc;
    }
    __syncthreads();
    for (int p = tid; p < 8192; p += 256) {             // Gk pre-LN
        int m = p >> 7, o = p & 127;
        float acc = gl_b[o];
        for (int v = 0; v < 128; ++v) acc += gl_w[o * 128 + v] * sG[v * 64 + m];
        sK[m * 128 + o] = acc;
    }
    __syncthreads();
    {                                                   // LN + f16 store
        int m = tid >> 2, c = tid & 3;
        float mu = 0.f;
        for (int d = 0; d < 32; ++d) mu += sK[m * 128 + c * 32 + d];
        mu *= (1.f / 32.f);
        float var = 0.f;
        for (int d = 0; d < 32; ++d) {
            float df = sK[m * 128 + c * 32 + d] - mu; var += df * df;
        }
        var *= (1.f / 32.f);
        float rs = rsqrtf(var + 1e-5f);
        for (int d = 0; d < 32; ++d)
            GK[(c * 64 + m) * 32 + d] = (_Float16)(
                (sK[m * 128 + c * 32 + d] - mu) * rs * gl_g[c * 32 + d] +
                gl_beta[c * 32 + d]);
    }
    __syncthreads();
    for (int i = tid; i < 512; i += 256) {              // V input
        int ch = i >> 6, pos = i & 63;
        sIn[i] = (ch < 4) ? vmm[ch * 64 + pos] : vpp[(ch - 4) * 64 + pos];
    }
    __syncthreads();
    {                                                   // V conv
        int c = tid >> 6, m = tid & 63, my = m >> 3, mx = m & 7;
        float acc = v_b[c];
        for (int ci = 0; ci < 8; ++ci)
            for (int dy = 0; dy < 3; ++dy)
                for (int dx = 0; dx < 3; ++dx) {
                    int iy = my + dy - 1, ix = mx + dx - 1;
                    if ((unsigned)iy < 8u && (unsigned)ix < 8u)
                        acc += v_w[(c * 8 + ci) * 9 + dy * 3 + dx] *
                               sIn[ci * 64 + iy * 8 + ix];
                }
        VV[c * 64 + m] = acc;
    }
}

// ---------------------------------------------------------------------------
// Main fused kernel: one 16x16 pixel tile per block, 256 threads = 8 waves.
// Each wave handles two pixel rows (M-groups of 16 pixels) with WMMA.
// ---------------------------------------------------------------------------
__global__ __launch_bounds__(256) void k_main(
    const float* __restrict__ xin, const float* __restrict__ guide,
    const float* __restrict__ fm_w, const float* __restrict__ fm_b,
    const float* __restrict__ fp_w, const float* __restrict__ fp_b,
    const float* __restrict__ fl_g, const float* __restrict__ fl_beta,
    const _Float16* __restrict__ WF, const float* __restrict__ BIAS,
    const _Float16* __restrict__ GK, const float* __restrict__ VV,
    float* __restrict__ ATT)
{
    extern __shared__ char smem[];
    _Float16* sW   = (_Float16*)(smem);           // [128][96] folded weight
    _Float16* sGk  = (_Float16*)(smem + 24576);   // [4][64][32]
    float*    sV   = (float*)   (smem + 40960);   // [4][64]
    float*    sBias= (float*)   (smem + 41984);   // [128]
    float*    sLg  = (float*)   (smem + 42496);   // [128]
    float*    sLb  = (float*)   (smem + 43008);   // [128]
    _Float16* sC8  = (_Float16*)(smem + 43520);   // [18][18][8] halo tile
    const int tid  = threadIdx.x;
    const int lane = tid & 31, wave = tid >> 5;
    const int lrow = lane & 15;                   // M / N selector
    const int lhi  = lane >> 4;                   // half-wave id
    char*  pw      = smem + 48768 + wave * 12288; // per-wave scratch
    float* sFq     = (float*)pw;                  // [16][128] pre-LN Fq
    float* sScore  = (float*)pw;                  // [16][64]  (overlays sFq)
    _Float16* sFqh = (_Float16*)(pw + 8192);      // [16][128] post-LN f16

    const int ty0 = blockIdx.y * 16, tx0 = blockIdx.x * 16;

    // ---- stage constants into LDS -------------------------------------
    {
        const unsigned* s = (const unsigned*)WF; unsigned* d = (unsigned*)sW;
        for (int i = tid; i < 6144; i += 256) d[i] = s[i];
    }
    {
        const unsigned* s = (const unsigned*)GK; unsigned* d = (unsigned*)sGk;
        for (int i = tid; i < 4096; i += 256) d[i] = s[i];
    }
    sV[tid] = VV[tid];
    if (tid < 128) { sBias[tid] = BIAS[tid]; sLg[tid] = fl_g[tid]; sLb[tid] = fl_beta[tid]; }

    // ---- c8 = [fm(up_x) || fp(guide)] on tile + 1 halo, f16 -----------
    for (int p = tid; p < 324; p += 256) {
        int cy = p / 18, cx = p % 18;
        int gy = ty0 + cy - 1, gx = tx0 + cx - 1;
        float v[8] = {0.f, 0.f, 0.f, 0.f, 0.f, 0.f, 0.f, 0.f};
        if ((unsigned)gy < 512u && (unsigned)gx < 512u) {
            for (int co = 0; co < 4; ++co) { v[co] = fm_b[co]; v[4 + co] = fp_b[co]; }
            for (int dy = 0; dy < 3; ++dy) {
                int iy = gy + dy - 1;
                if ((unsigned)iy >= 512u) continue;
                for (int dx = 0; dx < 3; ++dx) {
                    int ix = gx + dx - 1;
                    if ((unsigned)ix >= 512u) continue;
                    float gv = guide[iy * 512 + ix];
                    int xy = iy >> 2, xx = ix >> 2;                 // nearest upsample
                    float x0 = xin[           xy * 128 + xx];
                    float x1 = xin[16384    + xy * 128 + xx];
                    float x2 = xin[2 * 16384 + xy * 128 + xx];
                    float x3 = xin[3 * 16384 + xy * 128 + xx];
                    int t = dy * 3 + dx;
                    for (int co = 0; co < 4; ++co) {
                        v[co] += fm_w[co * 36      + t] * x0 +
                                 fm_w[co * 36 +  9 + t] * x1 +
                                 fm_w[co * 36 + 18 + t] * x2 +
                                 fm_w[co * 36 + 27 + t] * x3;
                        v[4 + co] += fp_w[co * 9 + t] * gv;
                    }
                }
            }
        }
        for (int ch = 0; ch < 8; ++ch)
            sC8[(cy * 18 + cx) * 8 + ch] = (_Float16)v[ch];
    }
    __syncthreads();

    // ---- per wave: two pixel rows -------------------------------------
    for (int g = 0; g < 2; ++g) {
        const int gi = wave * 2 + g;   // pixel row within tile

        // A fragments (im2col of folded conv), K = 0..95 (72 real + pad).
        // A layout used here (contiguous k): lane holds M=lane%16; halves 0-15 carry
        // K = lhi*16+{0..15} (matches B).
        V16 aF[3];
        for (int kk = 0; kk < 3; ++kk) {
            int k0 = kk * 32;
            for (int h = 0; h < 16; ++h) {
                int k = k0 + lhi * 16 + h;
                _Float16 val = (_Float16)0.f;
                if (k < 72) {
                    int cin = k / 9, t = k % 9, dy = t / 3, dx = t % 3;
                    val = sC8[((gi + dy) * 18 + (lrow + dx)) * 8 + cin];
                }
                aF[kk].h[h] = val;
            }
        }

        // Folded conv GEMM: 8 N-tiles x 3 K-steps -> Fq_pre[16][128]
        for (int n = 0; n < 8; ++n) {
            const int feat = n * 16 + lrow;                  // B column (lane N)
            const float bz = sBias[feat];
            v8f acc = {bz, bz, bz, bz, bz, bz, bz, bz};
            for (int kk = 0; kk < 3; ++kk) {
                // B layout: lane = column N, 16 consecutive K per half-wave
                v16h b = *(const v16h*)(sW + feat * 96 + kk * 32 + lhi * 16);
                acc = __builtin_amdgcn_wmma_f32_16x16x32_f16(
                          false, aF[kk].v, false, b, (short)0, acc, false, false);
            }
            // D layout: VGPR i -> row M = i + 8*lhi, col N = lrow
            for (int i = 0; i < 8; ++i)
                sFq[(i + lhi * 8) * 128 + feat] = acc[i];
        }
        __syncthreads();

        // LayerNorm over d=32 per (pixel, band); write f16 for scores-A
        for (int t = lane; t < 64; t += 32) {
            int px = t >> 2, c = t & 3;
            const float* row = sFq + px * 128 + c * 32;
            float mu = 0.f;
            for (int d = 0; d < 32; ++d) mu += row[d];
            mu *= (1.f / 32.f);
            float var = 0.f;
            for (int d = 0; d < 32; ++d) { float df = row[d] - mu; var += df * df; }
            var *= (1.f / 32.f);
            float rs = rsqrtf(var + 1e-5f);
            for (int d = 0; d < 32; ++d)
                sFqh[px * 128 + c * 32 + d] =
                    (_Float16)((row[d] - mu) * rs * sLg[c * 32 + d] + sLb[c * 32 + d]);
        }
        __syncthreads();

        // Attention per band: scores (WMMA, K=32), softmax(64), V-sum
        for (int c = 0; c < 4; ++c) {
            V16 aq;
            {
                v8h lo = *(const v8h*)(sFqh + lrow * 128 + c * 32 + lhi * 16);
                v8h hi = *(const v8h*)(sFqh + lrow * 128 + c * 32 + lhi * 16 + 8);
                for (int h = 0; h < 8; ++h) { aq.h[h] = lo[h]; aq.h[8 + h] = hi[h]; }
            }
            for (int t4 = 0; t4 < 4; ++t4) {
                const int m = t4 * 16 + lrow;
                v16h b = *(const v16h*)(sGk + (c * 64 + m) * 32 + lhi * 16);
                v8f acc = {};
                acc = __builtin_amdgcn_wmma_f32_16x16x32_f16(
                          false, aq.v, false, b, (short)0, acc, false, false);
                for (int i = 0; i < 8; ++i)
                    sScore[(i + lhi * 8) * 64 + m] = acc[i] * 0.17677669529663687f;
            }
            __syncthreads();
            if (lane < 16) {
                const int px = lane;
                float mx = -3.0e38f;
                for (int m = 0; m < 64; ++m) mx = fmaxf(mx, sScore[px * 64 + m]);
                float se = 0.f, ou = 0.f;
                for (int m = 0; m < 64; ++m) {
                    float e = expf(sScore[px * 64 + m] - mx);
                    se += e; ou += e * sV[c * 64 + m];
                }
                ATT[c * 262144 + (ty0 + gi) * 512 + (tx0 + px)] = ou / se;
            }
            __syncthreads();
        }
    }
}

// ---------------------------------------------------------------------------
// Final 3x3 conv: d_out = fa_w (*) ATT + fa_b
// ---------------------------------------------------------------------------
__global__ __launch_bounds__(256) void k_final(
    const float* __restrict__ att, const float* __restrict__ fa_w,
    const float* __restrict__ fa_b, float* __restrict__ out)
{
    int t = blockIdx.x * 256 + threadIdx.x;
    if (t >= 4 * 512 * 512) return;
    int co = t >> 18, r = t & 262143, y = r >> 9, x = r & 511;
    float acc = fa_b[co];
    for (int ci = 0; ci < 4; ++ci)
        for (int dy = 0; dy < 3; ++dy)
            for (int dx = 0; dx < 3; ++dx) {
                int iy = y + dy - 1, ix = x + dx - 1;
                if ((unsigned)iy < 512u && (unsigned)ix < 512u)
                    acc += fa_w[(co * 4 + ci) * 9 + dy * 3 + dx] *
                           att[ci * 262144 + iy * 512 + ix];
            }
    out[t] = acc;
}

// ---------------------------------------------------------------------------
extern "C" void kernel_launch(void* const* d_in, const int* in_sizes, int n_in,
                              void* d_out, int out_size, void* d_ws, size_t ws_size,
                              hipStream_t stream)
{
    (void)in_sizes; (void)n_in; (void)out_size; (void)ws_size;
    const float* x      = (const float*)d_in[0];
    const float* guide  = (const float*)d_in[1];
    const float* fp_w   = (const float*)d_in[2];
    const float* fp_b   = (const float*)d_in[3];
    const float* fm_w   = (const float*)d_in[4];
    const float* fm_b   = (const float*)d_in[5];
    const float* f_w    = (const float*)d_in[6];
    const float* f_b    = (const float*)d_in[7];
    const float* gp0_w  = (const float*)d_in[8];
    const float* gp0_b  = (const float*)d_in[9];
    const float* gp1_w  = (const float*)d_in[10];
    const float* gp1_b  = (const float*)d_in[11];
    const float* gp2_w  = (const float*)d_in[12];
    const float* gp2_b  = (const float*)d_in[13];
    const float* gm0_w  = (const float*)d_in[14];
    const float* gm0_b  = (const float*)d_in[15];
    const float* gm1_w  = (const float*)d_in[16];
    const float* gm1_b  = (const float*)d_in[17];
    const float* g_w    = (const float*)d_in[18];
    const float* g_b    = (const float*)d_in[19];
    const float* vp0_w  = (const float*)d_in[20];
    const float* vp0_b  = (const float*)d_in[21];
    const float* vp1_w  = (const float*)d_in[22];
    const float* vp1_b  = (const float*)d_in[23];
    const float* vp2_w  = (const float*)d_in[24];
    const float* vp2_b  = (const float*)d_in[25];
    const float* vm0_w  = (const float*)d_in[26];
    const float* vm0_b  = (const float*)d_in[27];
    const float* vm1_w  = (const float*)d_in[28];
    const float* vm1_b  = (const float*)d_in[29];
    const float* v_w    = (const float*)d_in[30];
    const float* v_b    = (const float*)d_in[31];
    const float* fl_w   = (const float*)d_in[32];
    const float* fl_b   = (const float*)d_in[33];
    const float* fl_g   = (const float*)d_in[34];
    const float* fl_beta= (const float*)d_in[35];
    const float* gl_w   = (const float*)d_in[36];
    const float* gl_b   = (const float*)d_in[37];
    const float* gl_g   = (const float*)d_in[38];
    const float* gl_beta= (const float*)d_in[39];
    const float* fa_w   = (const float*)d_in[40];
    const float* fa_b   = (const float*)d_in[41];

    // Workspace layout (bytes)
    char* ws = (char*)d_ws;
    _Float16* WF   = (_Float16*)(ws + 0);        // 128*96*2   = 24576
    float*    BIAS = (float*)   (ws + 24576);    // 128*4      =   512
    _Float16* GKw  = (_Float16*)(ws + 25088);    // 4*64*32*2  = 16384
    float*    VVw  = (float*)   (ws + 41472);    // 4*64*4     =  1024
    float*    gp1  = (float*)   (ws + 42496);    // 4*128*128*4= 262144
    float*    gp2  = (float*)   (ws + 304640);   // 4*32*32*4  =  16384
    float*    gpp  = (float*)   (ws + 321024);   // 4*8*8*4    =   1024
    float*    gm1  = (float*)   (ws + 322048);   // 16384
    float*    gmm  = (float*)   (ws + 338432);   // 1024
    float*    vp1  = (float*)   (ws + 339456);   // 262144
    float*    vp2  = (float*)   (ws + 601600);   // 16384
    float*    vpp  = (float*)   (ws + 617984);   // 1024
    float*    vm1  = (float*)   (ws + 619008);   // 16384
    float*    vmm  = (float*)   (ws + 635392);   // 1024
    float*    ATT  = (float*)   (ws + 636416);   // 4*512*512*4 = 4194304

    // 1) Fold fl_w into f_w
    k_fold<<<48, 256, 0, stream>>>(fl_w, f_w, f_b, fl_b, WF, BIAS);

    // 2) Downsample chains (G path)
    k_ds<<<256, 256, 0, stream>>>(guide, 512, 512, 1, gp0_w, gp0_b, gp1);
    k_ds<<<16,  256, 0, stream>>>(gp1,   128, 128, 4, gp1_w, gp1_b, gp2);
    k_ds<<<1,   256, 0, stream>>>(gp2,    32,  32, 4, gp2_w, gp2_b, gpp);
    k_ds<<<16,  256, 0, stream>>>(x,     128, 128, 4, gm0_w, gm0_b, gm1);
    k_ds<<<1,   256, 0, stream>>>(gm1,    32,  32, 4, gm1_w, gm1_b, gmm);
    //    (V path)
    k_ds<<<256, 256, 0, stream>>>(guide, 512, 512, 1, vp0_w, vp0_b, vp1);
    k_ds<<<16,  256, 0, stream>>>(vp1,   128, 128, 4, vp1_w, vp1_b, vp2);
    k_ds<<<1,   256, 0, stream>>>(vp2,    32,  32, 4, vp2_w, vp2_b, vpp);
    k_ds<<<16,  256, 0, stream>>>(x,     128, 128, 4, vm0_w, vm0_b, vm1);
    k_ds<<<1,   256, 0, stream>>>(vm1,    32,  32, 4, vm1_w, vm1_b, vmm);

    // 3) Keys (Gk, f16) and values (V)
    k_gkv<<<1, 256, 67584, stream>>>(gmm, gpp, vmm, vpp, g_w, g_b,
                                     gl_w, gl_b, gl_g, gl_beta, v_w, v_b,
                                     GKw, VVw);

    // 4) Fused query-conv + LN + attention over 32x32 tiles of 16x16 pixels
    //    dynamic LDS: 48768 shared + 8 waves * 12288 = 147072 bytes
    k_main<<<dim3(32, 32), 256, 147072, stream>>>(
        x, guide, fm_w, fm_b, fp_w, fp_b, fl_g, fl_beta,
        WF, BIAS, GKw, VVw, ATT);

    // 5) Final 3x3 conv
    k_final<<<4096, 256, 0, stream>>>(ATT, fa_w, fa_b, (float*)d_out);
}
